// ComplexLSTMClassifier_33449205301309
// MI455X (gfx1250) — compile-verified, matrix operand order checked by reading, not run on hardware
//
#include <hip/hip_runtime.h>
#include <hip/hip_bf16.h>

// ---------------------------------------------------------------------------
// ComplexLSTMClassifier for MI455X (gfx1250, wave32, WMMA)
//   Phase 1: pre0 = x @ W_ih0^T + b0   (HBM-bound; bf16 hi/lo split WMMA)
//   Phase 2: 4-layer LSTM systolic pipeline, 1 wave per layer, f16 WMMA
//            with fused input+recurrent matmul (K=32), plus MLP head.
//   Async global->LDS staging (ASYNCcnt) where the toolchain supports it.
// ---------------------------------------------------------------------------

typedef __attribute__((ext_vector_type(16))) __bf16   v16bf;
typedef __attribute__((ext_vector_type(16))) _Float16 v16h;
typedef __attribute__((ext_vector_type(8)))  float    v8f;

constexpr int kI     = 1086;   // input feature dim
constexpr int kKpad  = 1088;   // padded to multiple of 32
constexpr int kNch   = kKpad / 32;
constexpr int kH     = 16;     // hidden
constexpr int kG     = 64;     // 4*H gates
constexpr int kT     = 200;    // timesteps
constexpr int kB     = 256;    // batch
constexpr int kRows  = kB * kT; // 51200 flattened (b*T + t)
constexpr int kC     = 15;     // classes

// ---- optional gfx1250 async global->LDS path (toolchain-dependent) --------
#if defined(__has_builtin)
#  if __has_builtin(__builtin_amdgcn_global_load_async_to_lds_b128) && \
      __has_builtin(__builtin_amdgcn_s_wait_asynccnt)
#    define USE_ASYNC_LDS 1
#  endif
#endif
#ifndef USE_ASYNC_LDS
#  define USE_ASYNC_LDS 0
#endif

#if USE_ASYNC_LDS
// Builtin signature (from hipcc diagnostic): param 0 is
// 'int __attribute__((vector_size(16))) __device__ *' (addrspace(1));
// param 1 is the LDS-side (addrspace(3)) counterpart.
typedef int v4i_gcc __attribute__((vector_size(16)));
typedef __attribute__((address_space(1))) v4i_gcc* gv4i_ptr;
typedef __attribute__((address_space(3))) v4i_gcc* lv4i_ptr;
// AS1 == flat address; AS3 offset == low 32 bits of the flat shared pointer
// (ISA: LDS aperture maps via addr[31:0]).
static __device__ __forceinline__ void async_g2l_b128(const void* g, void* l) {
  __builtin_amdgcn_global_load_async_to_lds_b128(
      (gv4i_ptr)(unsigned long long)g,
      (lv4i_ptr)(unsigned)(unsigned long long)l, 0, 0);
}
#endif

static __device__ __forceinline__ float bf2f(__bf16 b) {
  unsigned short s = __builtin_bit_cast(unsigned short, b);
  unsigned u = ((unsigned)s) << 16;
  return __builtin_bit_cast(float, u);
}
static __device__ __forceinline__ __bf16 f2bf_rn(float f) {
  unsigned u = __builtin_bit_cast(unsigned, f);
  u += 0x7FFFu + ((u >> 16) & 1u);   // round-to-nearest-even on bf16 boundary
  unsigned short s = (unsigned short)(u >> 16);
  return __builtin_bit_cast(__bf16, s);
}
static __device__ __forceinline__ float sigm(float x) {
  return 1.f / (1.f + __expf(-x));
}
static __device__ __forceinline__ float tanh_fast(float x) {
  return 2.f / (1.f + __expf(-2.f * x)) - 1.f;
}

// ---------------------------------------------------------------------------
// Kernel 1: split W_ih0 (64 x 1086 f32) into bf16 hi/lo, padded to 1088.
// ---------------------------------------------------------------------------
__global__ void prep_w0(const float* __restrict__ w,
                        __bf16* __restrict__ whi, __bf16* __restrict__ wlo) {
  int idx = blockIdx.x * 256 + threadIdx.x;
  if (idx >= kG * kKpad) return;
  int g = idx / kKpad, k = idx % kKpad;
  float v = (k < kI) ? w[g * kI + k] : 0.f;
  __bf16 hi = f2bf_rn(v);
  whi[idx] = hi;
  wlo[idx] = f2bf_rn(v - bf2f(hi));
}

// ---------------------------------------------------------------------------
// Kernel 2: pre0[row][gate] = sum_k x[row][k]*W[gate][k] + b[gate]
//   grid = 400 blocks, 256 threads (8 waves). Block tile: 128 rows x 64 gates.
//   bf16 hi/lo split: 3 WMMAs per 16x16 tile per K=32 chunk.
//   W tiles staged via async global->LDS, overlapped with x conversion VALU.
// ---------------------------------------------------------------------------
__global__ __launch_bounds__(256)
void pre_gemm(const float* __restrict__ x,
              const __bf16* __restrict__ whi_g, const __bf16* __restrict__ wlo_g,
              const float* __restrict__ bias, float* __restrict__ pre) {
  __shared__ __bf16 xhi[128][32];
  __shared__ __bf16 xlo[128][32];
  __shared__ __bf16 whs[64][32];
  __shared__ __bf16 wls[64][32];

  const int tid  = threadIdx.x;
  const int lane = tid & 31;
  const int wv   = tid >> 5;          // wave 0..7 -> rows [16w,16w+16)
  const int rowBase = blockIdx.x * 128;
  const int n    = lane & 15;         // N / M position within 16
  const int half = lane >> 4;         // K-half / row-half selector

  v8f acc[4];
  for (int t = 0; t < 4; ++t) {
    float b = bias[t * 16 + n];
    for (int r = 0; r < 8; ++r) acc[t][r] = b;
  }

  const int kcol = tid & 31;          // staging: K column this thread loads
  const int rgrp = tid >> 5;          // staging: row group

  for (int kc = 0; kc < kNch; ++kc) {
    const int kbase = kc * 32;

#if USE_ASYNC_LDS
    // stage W tile (64 x 32 bf16, hi+lo) as 512 async b128 copies (2/thread),
    // overlapped with the VALU-heavy x split below.
    for (int i = 0; i < 2; ++i) {
      int idx = tid + 256 * i;        // 0..511
      int arr = idx >> 8;             // 0 = hi, 1 = lo
      int j   = idx & 255;
      int g   = j >> 2;               // gate row 0..63
      int q   = j & 3;                // 16B segment (8 bf16)
      const __bf16* src = (arr ? wlo_g : whi_g) + g * kKpad + kbase + q * 8;
      __bf16* dst = arr ? &wls[g][q * 8] : &whs[g][q * 8];
      async_g2l_b128(src, dst);
    }
#else
    for (int i = 0; i < 8; ++i) {
      int g = rgrp + 8 * i;
      whs[g][kcol] = whi_g[g * kKpad + kbase + kcol];
      wls[g][kcol] = wlo_g[g * kKpad + kbase + kcol];
    }
#endif

    // stage x tile (128 x 32) -> bf16 hi/lo in LDS (coalesced: lanes vary K)
    for (int i = 0; i < 16; ++i) {
      int rl = rgrp + 8 * i;
      int kg = kbase + kcol;
      float v = (kg < kI) ? x[(size_t)(rowBase + rl) * kI + kg] : 0.f;
      __bf16 hi = f2bf_rn(v);
      xhi[rl][kcol] = hi;
      xlo[rl][kcol] = f2bf_rn(v - bf2f(hi));
    }
    if (kc + 1 < kNch)   // gfx1250 global_prefetch_b8 for the next x chunk
      __builtin_prefetch(&x[(size_t)(rowBase + rgrp) * kI + kbase + 32 + kcol], 0, 1);

#if USE_ASYNC_LDS
    __builtin_amdgcn_s_wait_asynccnt(0);
#endif
    __syncthreads();

    const int arow = wv * 16 + n;
    v16bf ahi = *(const v16bf*)&xhi[arow][half * 16];
    v16bf alo = *(const v16bf*)&xlo[arow][half * 16];
    for (int t = 0; t < 4; ++t) {
      const int g = t * 16 + n;
      v16bf bhi = *(const v16bf*)&whs[g][half * 16];
      v16bf blo = *(const v16bf*)&wls[g][half * 16];
      acc[t] = __builtin_amdgcn_wmma_f32_16x16x32_bf16(false, ahi, false, bhi,
                                                       (short)0, acc[t], false, false);
      acc[t] = __builtin_amdgcn_wmma_f32_16x16x32_bf16(false, alo, false, bhi,
                                                       (short)0, acc[t], false, false);
      acc[t] = __builtin_amdgcn_wmma_f32_16x16x32_bf16(false, ahi, false, blo,
                                                       (short)0, acc[t], false, false);
    }
    __syncthreads();
  }

  for (int t = 0; t < 4; ++t)
    for (int r = 0; r < 8; ++r) {
      int row = rowBase + wv * 16 + half * 8 + r;
      pre[(size_t)row * kG + t * 16 + n] = acc[t][r];
    }
}

// ---------------------------------------------------------------------------
// Kernel 3: fused 4-layer LSTM recurrence + MLP head + softmax.
//   grid = 16 blocks (16 batch rows each), 128 threads = 4 waves.
//   Wave L == layer L; skewed systolic pipeline over time, LDS h exchange.
//   Per layer-step: 4x v_wmma_f32_16x16x32_f16, A = [x_in(K0..15)|h_prev(K16..31)].
//   Wave 0's gate pre-activations prefetched one step ahead via async->LDS.
// ---------------------------------------------------------------------------
__global__ __launch_bounds__(128)
void lstm_fused(const float* __restrict__ pre,
                const float* __restrict__ whh0,
                const float* __restrict__ wih1, const float* __restrict__ whh1,
                const float* __restrict__ b1,
                const float* __restrict__ wih2, const float* __restrict__ whh2,
                const float* __restrict__ b2,
                const float* __restrict__ wih3, const float* __restrict__ whh3,
                const float* __restrict__ b3,
                const float* __restrict__ wfc1, const float* __restrict__ bfc1,
                const float* __restrict__ wfc2, const float* __restrict__ bfc2,
                float* __restrict__ out) {
  __shared__ float hbuf[4][2][16][16];   // [layer][parity][batch][hidden]
  __shared__ float hfin[16][16];
  __shared__ float hfc[16][16];
  __shared__ float logit[16][kC];
#if USE_ASYNC_LDS
  __shared__ float pbuf[2][16][64];      // double-buffered gate pre-activations
#endif

  const int tid   = threadIdx.x;
  const int lane  = tid & 31;
  const int layer = tid >> 5;           // wave index == LSTM layer
  const int bg    = blockIdx.x * 16;    // batch group base
  const int n     = lane & 15;
  const int half  = lane >> 4;

  for (int i = tid; i < 4 * 2 * 16 * 16; i += 128) ((float*)hbuf)[i] = 0.f;

  const float* wih = (layer == 1) ? wih1 : (layer == 2) ? wih2 : wih3;
  const float* whh = (layer == 0) ? whh0 : (layer == 1) ? whh1
                                         : (layer == 2) ? whh2 : whh3;
  const float* bb  = (layer == 1) ? b1 : (layer == 2) ? b2 : b3;

  // B fragments: K<16 -> W_ih (layer0: W_hh), K>=16 -> W_hh (layer0: 0)
  v16h  bfrag[4];
  float cbias[4];
  for (int t = 0; t < 4; ++t) {
    int g = t * 16 + n;
    for (int e = 0; e < 16; ++e) {
      int kidx = half * 16 + e;
      float v;
      if (layer == 0) v = (kidx < 16) ? whh[g * kH + kidx] : 0.f;
      else            v = (kidx < 16) ? wih[g * kH + kidx]
                                      : whh[g * kH + (kidx - 16)];
      bfrag[t][e] = (_Float16)v;
    }
    cbias[t] = (layer == 0) ? 0.f : bb[g];
  }

  v8f cst;                              // cell state, D-layout, f32 in regs
  for (int r = 0; r < 8; ++r) cst[r] = 0.f;

#if USE_ASYNC_LDS
  // wave 0: async-copy the 16x64 gate tile for timestep tt into pbuf[tt&1]
  // (256 b128 copies -> 8 per lane; rows perfectly 256B-coalesced)
  auto issue_pre = [&](int tt) {
    for (int i = 0; i < 8; ++i) {
      int idx = lane + 32 * i;          // 0..255
      int mb  = idx >> 4;               // batch row 0..15
      int q   = idx & 15;               // 16B segment within the 64 gates
      const float* src = &pre[((size_t)(bg + mb) * kT + tt) * kG + q * 4];
      async_g2l_b128(src, &pbuf[tt & 1][mb][q * 4]);
    }
  };
  if (layer == 0) issue_pre(0);
#endif

  __syncthreads();

  for (int k = 0; k < kT + 3; ++k) {    // skewed pipeline: t = k - layer
    const int t = k - layer;
    if (t >= 0 && t < kT) {             // wave-uniform predicate (EXEC all-1)
      const int pb = (k - 1) & 1;       // read parity (written last iteration)
      v16h a;
      if (half == 0) {                  // K 0..15: x_in (layer>=1) / h_prev (layer0)
        const float* lo = (layer == 0) ? &hbuf[0][pb][n][0]
                                       : &hbuf[layer - 1][pb][n][0];
        for (int e = 0; e < 16; ++e) a[e] = (_Float16)lo[e];
      } else {                          // K 16..31: own h_prev (layer>=1) / zeros
        if (layer == 0) {
          for (int e = 0; e < 16; ++e) a[e] = (_Float16)0.f;
        } else {
          const float* hp = &hbuf[layer][pb][n][0];
          for (int e = 0; e < 16; ++e) a[e] = (_Float16)hp[e];
        }
      }

      v8f gi, gf, gg, go;
      if (layer == 0) {                 // C init = precomputed input projection
#if USE_ASYNC_LDS
        if (t + 1 < kT) {               // prefetch next step, wait for this one
          issue_pre(t + 1);
          __builtin_amdgcn_s_wait_asynccnt(8);
        } else {
          __builtin_amdgcn_s_wait_asynccnt(0);
        }
        for (int r = 0; r < 8; ++r) {
          int mb = half * 8 + r;
          gi[r] = pbuf[t & 1][mb][ 0 + n];
          gf[r] = pbuf[t & 1][mb][16 + n];
          gg[r] = pbuf[t & 1][mb][32 + n];
          go[r] = pbuf[t & 1][mb][48 + n];
        }
#else
        for (int r = 0; r < 8; ++r) {
          size_t row = (size_t)(bg + half * 8 + r) * kT + t;
          gi[r] = pre[row * kG +  0 + n];
          gf[r] = pre[row * kG + 16 + n];
          gg[r] = pre[row * kG + 32 + n];
          go[r] = pre[row * kG + 48 + n];
        }
#endif
      } else {                          // C init = bias (input proj fused in WMMA)
        for (int r = 0; r < 8; ++r) {
          gi[r] = cbias[0]; gf[r] = cbias[1]; gg[r] = cbias[2]; go[r] = cbias[3];
        }
      }
      gi = __builtin_amdgcn_wmma_f32_16x16x32_f16(false, a, false, bfrag[0],
                                                  (short)0, gi, false, false);
      gf = __builtin_amdgcn_wmma_f32_16x16x32_f16(false, a, false, bfrag[1],
                                                  (short)0, gf, false, false);
      gg = __builtin_amdgcn_wmma_f32_16x16x32_f16(false, a, false, bfrag[2],
                                                  (short)0, gg, false, false);
      go = __builtin_amdgcn_wmma_f32_16x16x32_f16(false, a, false, bfrag[3],
                                                  (short)0, go, false, false);

      const int wb = k & 1;             // write parity
      for (int r = 0; r < 8; ++r) {
        float iv = sigm(gi[r]);
        float fv = sigm(gf[r]);
        float gv = tanh_fast(gg[r]);
        float ov = sigm(go[r]);
        float c  = fv * cst[r] + iv * gv;
        cst[r]   = c;
        float hv = ov * tanh_fast(c);
        int mb = half * 8 + r;
        hbuf[layer][wb][mb][n] = hv;
        if (layer == 3 && t == kT - 1) hfin[mb][n] = hv;
      }
    }
    __syncthreads();
  }

  // ---- MLP head: relu(h @ wfc1^T + b1) @ wfc2^T + b2, softmax over 15 ----
  for (int idx = tid; idx < 16 * 16; idx += 128) {
    int row = idx >> 4, j = idx & 15;
    float s = bfc1[j];
    for (int kk = 0; kk < 16; ++kk) s += hfin[row][kk] * wfc1[j * 16 + kk];
    hfc[row][j] = fmaxf(s, 0.f);
  }
  __syncthreads();
  for (int idx = tid; idx < 16 * kC; idx += 128) {
    int row = idx / kC, c = idx % kC;
    float s = bfc2[c];
    for (int kk = 0; kk < 16; ++kk) s += hfc[row][kk] * wfc2[c * 16 + kk];
    logit[row][c] = s;
  }
  __syncthreads();
  if (tid < 16) {
    int row = tid;
    float mx = -3.4e38f;
    for (int c = 0; c < kC; ++c) mx = fmaxf(mx, logit[row][c]);
    float s = 0.f, ex[kC];
    for (int c = 0; c < kC; ++c) { ex[c] = __expf(logit[row][c] - mx); s += ex[c]; }
    float inv = 1.f / s;
    for (int c = 0; c < kC; ++c) out[(size_t)(bg + row) * kC + c] = ex[c] * inv;
  }
}

// ---------------------------------------------------------------------------
extern "C" void kernel_launch(void* const* d_in, const int* in_sizes, int n_in,
                              void* d_out, int out_size, void* d_ws, size_t ws_size,
                              hipStream_t stream) {
  const float* x    = (const float*)d_in[0];
  const float* wih0 = (const float*)d_in[1];
  const float* whh0 = (const float*)d_in[2];
  const float* b0   = (const float*)d_in[3];
  const float* wih1 = (const float*)d_in[4];
  const float* whh1 = (const float*)d_in[5];
  const float* b1   = (const float*)d_in[6];
  const float* wih2 = (const float*)d_in[7];
  const float* whh2 = (const float*)d_in[8];
  const float* b2   = (const float*)d_in[9];
  const float* wih3 = (const float*)d_in[10];
  const float* whh3 = (const float*)d_in[11];
  const float* b3   = (const float*)d_in[12];
  const float* wfc1 = (const float*)d_in[13];
  const float* bfc1 = (const float*)d_in[14];
  const float* wfc2 = (const float*)d_in[15];
  const float* bfc2 = (const float*)d_in[16];
  float* out = (float*)d_out;

  // workspace: [pre0 f32 51200x64 | W0 hi bf16 64x1088 | W0 lo bf16 64x1088]
  float*  pre = (float*)d_ws;
  __bf16* whi = (__bf16*)((char*)d_ws + (size_t)kRows * kG * sizeof(float));
  __bf16* wlo = whi + kG * kKpad;

  prep_w0  <<<(kG * kKpad + 255) / 256, 256, 0, stream>>>(wih0, whi, wlo);
  pre_gemm <<<kRows / 128, 256, 0, stream>>>(x, whi, wlo, b0, pre);
  lstm_fused<<<kB / 16, 128, 0, stream>>>(pre, whh0, wih1, whh1, b1,
                                          wih2, whh2, b2, wih3, whh3, b3,
                                          wfc1, bfc1, wfc2, bfc2, out);
}
